// Ansatz_fb_81879256531307
// MI455X (gfx1250) — matrix operand order repaired and checked by reading, not compiled, not run on hardware
//
#include <hip/hip_runtime.h>
#include <hip/hip_bf16.h>

// ---------------- types ----------------
typedef __attribute__((ext_vector_type(16))) _Float16 v16h;
typedef __attribute__((ext_vector_type(8)))  _Float16 v8h;
typedef __attribute__((ext_vector_type(8)))  float    v8f;

// ---------------- problem constants ----------------
#define NB   2048
#define NE   20
#define NU   10
#define NA   4
#define NDET 16
#define NSV  256
#define NPV  32
#define NSVO 832            // 3*256 + 2*32
#define D0P  64             // 56 padded to 64 (multiple of 32 for WMMA K)

// ---------------- workspace layout (bytes, 256-aligned) ----------------
static const size_t OFF_W16   = 0;                 // converted f16 weights, 2,332,672 B used
static const size_t OFF_SFEAT = 2359296;           // [B*20*16]    f16
static const size_t OFF_EXPV  = 3670016;           // [B*20]       f32
static const size_t OFF_PFEAT = 3833856;           // [B*20*20*4]  f16
static const size_t OFF_PV    = 10387456;          // [B*20*20*32] f16
static const size_t OFF_PMEAN = 62816256;          // [B*20*64]    f16 (u|d)
static const size_t OFF_SMEAN = 68059136;          // [B*512]      f16 (u|d)
static const size_t OFF_BLK0  = 70156288;          // [B*20*64]    f16 (layer-0 concat only)
static const size_t OFF_SV    = 75399168;          // [B*20*256]   f16 (reused for s_u/s_d)
static const size_t OFF_ORB   = 96370688;          // [2][B*16][100] f32
static const size_t OFF_LD    = 122585088;         // [2][B*16]    f32
static const size_t OFF_SGN   = 122847232;         // [2][B*16]    f32

// sub-offsets inside W16 region (in halfs)
static const size_t W_S0T = 0;          // 256*64
static const size_t W_ST  = 16384;      // 3 * 256*832
static const size_t W_VUT = 655360;     // 256*832
static const size_t W_VDT = 868352;     // 256*832
static const size_t W_WUT = 1081344;    // 160*256
static const size_t W_WDT = 1122304;    // 160*256
static const size_t W_PT  = 1163264;    // 3 * 32*32

// ================= weight convert: f32 [K,N] -> f16 [N,Kpad] (zero padded) =================
__global__ void cvt_w_kernel(const float* __restrict__ src, _Float16* __restrict__ dst,
                             int K, int N, int Kpad) {
  int idx = blockIdx.x * 256 + threadIdx.x;
  if (idx >= N * Kpad) return;
  int n = idx / Kpad, k = idx % Kpad;
  dst[idx] = (k < K) ? (_Float16)src[(size_t)k * N + n] : (_Float16)0.f;
}

// ================= electron-atom features + exponential envelope =================
__global__ void feat_kernel(const float* __restrict__ r, const float* __restrict__ a,
                            _Float16* __restrict__ sfeat, float* __restrict__ expv) {
  int idx = blockIdx.x * 256 + threadIdx.x;          // (b, e)
  if (idx >= NB * NE) return;
  float rx = r[idx * 3 + 0], ry = r[idx * 3 + 1], rz = r[idx * 3 + 2];
  float es = 0.f;
#pragma unroll
  for (int at = 0; at < NA; ++at) {
    float dx = rx - a[at * 3 + 0], dy = ry - a[at * 3 + 1], dz = rz - a[at * 3 + 2];
    float len = sqrtf(dx * dx + dy * dy + dz * dz);
    sfeat[(size_t)idx * 16 + at * 4 + 0] = (_Float16)dx;
    sfeat[(size_t)idx * 16 + at * 4 + 1] = (_Float16)dy;
    sfeat[(size_t)idx * 16 + at * 4 + 2] = (_Float16)dz;
    sfeat[(size_t)idx * 16 + at * 4 + 3] = (_Float16)len;
    es += expf(-len);
  }
  expv[idx] = es;
}

// ================= pair features + pair layer 0 (4 -> 32, tanh) =================
__global__ void pair0_kernel(const float* __restrict__ r, const float* __restrict__ p_w0,
                             const float* __restrict__ p_b0,
                             _Float16* __restrict__ pv, _Float16* __restrict__ pfeat) {
  int idx = blockIdx.x * 256 + threadIdx.x;          // (b, i, j)
  if (idx >= NB * NE * NE) return;
  int j = idx % NE; int bi = idx / NE; int i = bi % NE; int b = bi / NE;
  const float* ri = r + ((size_t)b * NE + i) * 3;
  const float* rj = r + ((size_t)b * NE + j) * 3;
  float dx = rj[0] - ri[0], dy = rj[1] - ri[1], dz = rj[2] - ri[2];
  float eye = (i == j) ? 1.f : 0.f;
  float ex = dx + eye, ey = dy + eye, ez = dz + eye;
  float len = sqrtf(ex * ex + ey * ey + ez * ez);
  float f[4] = {dx, dy, dz, len};
#pragma unroll
  for (int k = 0; k < 4; ++k) pfeat[(size_t)idx * 4 + k] = (_Float16)f[k];
#pragma unroll 4
  for (int c = 0; c < NPV; ++c) {
    float acc = p_b0[c];
#pragma unroll
    for (int k = 0; k < 4; ++k) acc += f[k] * p_w0[k * NPV + c];
    pv[(size_t)idx * NPV + c] = (_Float16)tanhf(acc);
  }
}

// ================= pair means over first/second 10 electrons (axis i) =================
__global__ void pmean_kernel(const _Float16* __restrict__ p, _Float16* __restrict__ pmean,
                             int dp, int total /* NB*NE*dp */) {
  int idx = blockIdx.x * 256 + threadIdx.x;
  if (idx >= total) return;
  int c = idx % dp; int bj = idx / dp; int b = bj / NE; int j = bj % NE;
  const _Float16* base = p + ((size_t)b * NE * NE + j) * dp + c;   // i stride = NE*dp
  float su = 0.f, sd = 0.f;
#pragma unroll
  for (int i = 0; i < NU; ++i)  su += (float)base[(size_t)i * NE * dp];
#pragma unroll
  for (int i = NU; i < NE; ++i) sd += (float)base[(size_t)i * NE * dp];
  pmean[(size_t)bj * 2 * dp + c]      = (_Float16)(su * 0.1f);
  pmean[(size_t)bj * 2 * dp + dp + c] = (_Float16)(sd * 0.1f);
}

// ================= single-stream spin means =================
__global__ void smean_kernel(const _Float16* __restrict__ s, _Float16* __restrict__ smean,
                             int ds, int total /* NB*ds */) {
  int idx = blockIdx.x * 256 + threadIdx.x;
  if (idx >= total) return;
  int b = idx / ds, c = idx % ds;
  const _Float16* base = s + ((size_t)b * NE) * ds + c;
  float su = 0.f, sd = 0.f;
#pragma unroll
  for (int e = 0; e < NU; ++e)  su += (float)base[(size_t)e * ds];
#pragma unroll
  for (int e = NU; e < NE; ++e) sd += (float)base[(size_t)e * ds];
  smean[(size_t)b * 2 * ds + c]      = (_Float16)(su * 0.1f);
  smean[(size_t)b * 2 * ds + ds + c] = (_Float16)(sd * 0.1f);
}

// ================= layer-0 concat (regions not 8-aligned -> materialize) =================
__global__ void blk0_build_kernel(const _Float16* __restrict__ s, const _Float16* __restrict__ smean,
                                  const _Float16* __restrict__ pmean, _Float16* __restrict__ blk,
                                  int total /* NB*NE*64 */) {
  int idx = blockIdx.x * 256 + threadIdx.x;
  if (idx >= total) return;
  int c = idx % D0P; int be = idx / D0P; int b = be / NE;
  float v;
  if (c < 16)       v = (float)s[(size_t)be * 16 + c];
  else if (c < 48)  v = (float)smean[(size_t)b * 32 + (c - 16)];   // u(16) | d(16)
  else if (c < 56)  v = (float)pmean[(size_t)be * 8 + (c - 48)];   // u(4)  | d(4)
  else              v = 0.f;
  blk[(size_t)idx] = (_Float16)v;
}

// ================= WMMA GEMM: out = tanh(A @ Wt^T + bias) (+ res) =================
// 4 waves/block; each wave computes a 16 x (16*NT) strip: A fragment loaded once per
// K-step, NT B fragments -> NT wmma. FUSED: A synthesized on the fly from
// [ sv(256) | smean_u(256) | smean_d(256) | pmean_u(32) | pmean_d(32) ]  (K = 832);
// region chosen by the *uniform* k so the select is scalar (smean [b][u|d] collapses
// regions 256..768 to one base).
// map10: logical row r -> physical row (r/10)*20 + eoff + r%10 (A and out/res).
template <int NT, bool FUSED>
__global__ void gemm_tanh_t(const _Float16* __restrict__ A, int lda,
                            const _Float16* __restrict__ svp, const _Float16* __restrict__ smean,
                            const _Float16* __restrict__ pmean,
                            const _Float16* __restrict__ Wt, int ldb,
                            const float* __restrict__ bias,
                            _Float16* __restrict__ out, int ldo,
                            const _Float16* __restrict__ res,
                            int N, int K, int map10, int eoff) {
  int lane = threadIdx.x & 31, wave = threadIdx.x >> 5;
  int wcol = (blockIdx.y * 4 + wave) * (16 * NT);
  int r0 = blockIdx.x * 16;
  int m = lane & 15, h = lane >> 4;

  int lr = r0 + m;
  int ar = map10 ? (lr / 10) * 20 + eoff + (lr % 10) : lr;
  int bb_ = ar / NE;
  const _Float16* Arow  = FUSED ? nullptr : A + (size_t)ar * lda + h * 8;
  const _Float16* Asv   = FUSED ? svp   + (size_t)ar  * 256 : nullptr;   // region bases (per lane)
  const _Float16* Asm   = FUSED ? smean + (size_t)bb_ * 512 : nullptr;
  const _Float16* Apm   = FUSED ? pmean + (size_t)ar  * 64  : nullptr;
  const _Float16* Brow  = Wt + (size_t)(wcol + m) * ldb + h * 8;

  int nact = 0;
#pragma unroll
  for (int t = 0; t < NT; ++t) nact += (wcol + t * 16 < N) ? 1 : 0;
  nact = __builtin_amdgcn_readfirstlane(nact);     // provably wave-uniform -> scalar branches

  v8f acc[NT];
#pragma unroll
  for (int t = 0; t < NT; ++t) acc[t] = (v8f){};

  if (nact > 0) {
    for (int k = 0; k < K; k += 32) {
      v8h alo, ahi;
      if (FUSED) {
        // all chunks of one K-step share a region (boundaries 256/768 are multiples of 32)
        const _Float16* base;
        int coff;
        if (k < 256)      { base = Asv; coff = k; }
        else if (k < 768) { base = Asm; coff = k - 256; }
        else              { base = Apm; coff = k - 768; }
        alo = *(const v8h*)(base + coff + h * 8);
        ahi = *(const v8h*)(base + coff + 16 + h * 8);
      } else {
        alo = *(const v8h*)(Arow + k);
        ahi = *(const v8h*)(Arow + k + 16);
      }
      v16h a;
#pragma unroll
      for (int t = 0; t < 8; ++t) { a[t] = alo[t]; a[8 + t] = ahi[t]; }
#pragma unroll
      for (int t = 0; t < NT; ++t) {
        if (t < nact) {
          const _Float16* Br = Brow + (size_t)t * 16 * ldb + k;
          v8h blo = *(const v8h*)(Br);
          v8h bhi = *(const v8h*)(Br + 16);
          v16h bf;
#pragma unroll
          for (int q = 0; q < 8; ++q) { bf[q] = blo[q]; bf[8 + q] = bhi[q]; }
          acc[t] = __builtin_amdgcn_wmma_f32_16x16x32_f16(false, a, false, bf, (short)0,
                                                          acc[t], false, false);
        }
      }
    }
  }
  __syncthreads();                                 // all A reads complete before in-place writes
  if (nact > 0) {
#pragma unroll
    for (int t = 0; t < NT; ++t) {
      if (t < nact) {
        int col = wcol + t * 16 + m;
        float bv = bias ? bias[col] : 0.f;
#pragma unroll
        for (int v = 0; v < 8; ++v) {
          int rr = r0 + v + 8 * h;
          int orow = map10 ? (rr / 10) * 20 + eoff + (rr % 10) : rr;
          float val = tanhf(acc[t][v] + bv);
          if (res) val += (float)res[(size_t)orow * ldo + col];
          out[(size_t)orow * ldo + col] = (_Float16)val;
        }
      }
    }
  }
}

// ================= orbital GEMM: orb[b][det][e][j] = (s@W+b)[det*10+j] * exp[b,eoff+e] =================
__global__ void gemm_orb_kernel(const _Float16* __restrict__ A, int lda,
                                const _Float16* __restrict__ Wt, int ldb,
                                const float* __restrict__ bias, const float* __restrict__ expv,
                                float* __restrict__ orb, int eoff, int N, int K) {
  int lane = threadIdx.x & 31, wave = threadIdx.x >> 5;
  int wcol = wave * 64;
  int r0 = blockIdx.x * 16;
  int m = lane & 15, h = lane >> 4;

  int lr = r0 + m;
  int ar = (lr / 10) * 20 + eoff + (lr % 10);
  const _Float16* Arow = A + (size_t)ar * lda + h * 8;
  const _Float16* Brow = Wt + (size_t)(wcol + m) * ldb + h * 8;

  int nact = 0;
#pragma unroll
  for (int t = 0; t < 4; ++t) nact += (wcol + t * 16 < N) ? 1 : 0;
  nact = __builtin_amdgcn_readfirstlane(nact);

  v8f acc[4];
#pragma unroll
  for (int t = 0; t < 4; ++t) acc[t] = (v8f){};

  if (nact > 0) {
    for (int k = 0; k < K; k += 32) {
      v8h alo = *(const v8h*)(Arow + k);
      v8h ahi = *(const v8h*)(Arow + k + 16);
      v16h a;
#pragma unroll
      for (int t = 0; t < 8; ++t) { a[t] = alo[t]; a[8 + t] = ahi[t]; }
#pragma unroll
      for (int t = 0; t < 4; ++t) {
        if (t < nact) {
          const _Float16* Br = Brow + (size_t)t * 16 * ldb + k;
          v8h blo = *(const v8h*)(Br);
          v8h bhi = *(const v8h*)(Br + 16);
          v16h bf;
#pragma unroll
          for (int q = 0; q < 8; ++q) { bf[q] = blo[q]; bf[8 + q] = bhi[q]; }
          acc[t] = __builtin_amdgcn_wmma_f32_16x16x32_f16(false, a, false, bf, (short)0,
                                                          acc[t], false, false);
        }
      }
    }
#pragma unroll
    for (int t = 0; t < 4; ++t) {
      if (t < nact) {
        int col = wcol + t * 16 + m;
        int det = col / NU, jj = col % NU;
        float bv = bias[col];
#pragma unroll
        for (int v = 0; v < 8; ++v) {
          int rr = r0 + v + 8 * h;
          int w = rr / 10, e = rr % 10;
          float ex = expv[(size_t)w * NE + eoff + e];
          orb[(((size_t)w * NDET + det) * NU + e) * NU + jj] = (acc[t][v] + bv) * ex;
        }
      }
    }
  }
}

// ================= per-(b,det,spin) 10x10 slogdet via partial-pivot LU =================
__global__ void det_kernel(const float* __restrict__ orb, float* __restrict__ ldv,
                           float* __restrict__ sgnv, int total /* 2*NB*NDET */) {
  int t = blockIdx.x * 128 + threadIdx.x;
  if (t >= total) return;
  const float* src = orb + (size_t)t * 100;
  float mtx[100];
#pragma unroll
  for (int i = 0; i < 100; ++i) mtx[i] = src[i];
  float sign = 1.f, lg = 0.f;
  for (int k = 0; k < 10; ++k) {
    int p = k; float mx = fabsf(mtx[k * 10 + k]);
    for (int rI = k + 1; rI < 10; ++rI) {
      float v = fabsf(mtx[rI * 10 + k]);
      if (v > mx) { mx = v; p = rI; }
    }
    if (p != k) {
      for (int c = 0; c < 10; ++c) { float tv = mtx[k*10+c]; mtx[k*10+c] = mtx[p*10+c]; mtx[p*10+c] = tv; }
      sign = -sign;
    }
    float d = mtx[k * 10 + k];
    if (d < 0.f) sign = -sign;
    lg += logf(fabsf(d) + 1e-30f);
    float inv = 1.f / d;
    for (int rI = k + 1; rI < 10; ++rI) {
      float f = mtx[rI * 10 + k] * inv;
      for (int c = k; c < 10; ++c) mtx[rI * 10 + c] -= f * mtx[k * 10 + c];
    }
  }
  ldv[t] = lg; sgnv[t] = sign;
}

// ================= signed logsumexp det combine =================
__global__ void final_kernel(const float* __restrict__ ldv, const float* __restrict__ sgnv,
                             const float* __restrict__ wf, float* __restrict__ out) {
  int b = blockIdx.x * 128 + threadIdx.x;
  if (b >= NB) return;
  float l[NDET], s[NDET];
  float mx = -3.0e38f;
#pragma unroll
  for (int i = 0; i < NDET; ++i) {
    l[i] = ldv[(size_t)b * NDET + i] + ldv[(size_t)NB * NDET + (size_t)b * NDET + i];
    s[i] = sgnv[(size_t)b * NDET + i] * sgnv[(size_t)NB * NDET + (size_t)b * NDET + i];
    mx = fmaxf(mx, l[i]);
  }
  float psi = 0.f;
#pragma unroll
  for (int i = 0; i < NDET; ++i) psi += s[i] * expf(l[i] - mx) * wf[i];
  out[b] = logf(fabsf(psi)) + mx;
}

// ================= host orchestration =================
extern "C" void kernel_launch(void* const* d_in, const int* in_sizes, int n_in,
                              void* d_out, int out_size, void* d_ws, size_t ws_size,
                              hipStream_t stream) {
  (void)in_sizes; (void)n_in; (void)out_size; (void)ws_size;
  const float* r    = (const float*)d_in[0];
  const float* a    = (const float*)d_in[1];
  const float* s_w0 = (const float*)d_in[2];
  const float* s_b0 = (const float*)d_in[3];
  const float* s_w  = (const float*)d_in[4];
  const float* s_b  = (const float*)d_in[5];
  const float* p_w0 = (const float*)d_in[6];
  const float* p_b0 = (const float*)d_in[7];
  const float* p_w  = (const float*)d_in[8];
  const float* p_b  = (const float*)d_in[9];
  const float* vu_w = (const float*)d_in[10];
  const float* vu_b = (const float*)d_in[11];
  const float* vd_w = (const float*)d_in[12];
  const float* vd_b = (const float*)d_in[13];
  const float* wu_w = (const float*)d_in[14];
  const float* wu_b = (const float*)d_in[15];
  const float* wd_w = (const float*)d_in[16];
  const float* wd_b = (const float*)d_in[17];
  const float* wf_w = (const float*)d_in[18];
  float* out = (float*)d_out;

  char* ws = (char*)d_ws;
  _Float16* w16   = (_Float16*)(ws + OFF_W16);
  _Float16* sfeat = (_Float16*)(ws + OFF_SFEAT);
  float*    expv  = (float*)   (ws + OFF_EXPV);
  _Float16* pfeat = (_Float16*)(ws + OFF_PFEAT);
  _Float16* pv    = (_Float16*)(ws + OFF_PV);
  _Float16* pmean = (_Float16*)(ws + OFF_PMEAN);
  _Float16* smean = (_Float16*)(ws + OFF_SMEAN);
  _Float16* blk0  = (_Float16*)(ws + OFF_BLK0);
  _Float16* sv    = (_Float16*)(ws + OFF_SV);
  float*    orb   = (float*)   (ws + OFF_ORB);
  float*    ldv   = (float*)   (ws + OFF_LD);
  float*    sgnv  = (float*)   (ws + OFF_SGN);

  _Float16* s_w0t = w16 + W_S0T;
  _Float16* s_wt  = w16 + W_ST;
  _Float16* vut   = w16 + W_VUT;
  _Float16* vdt   = w16 + W_VDT;
  _Float16* wut   = w16 + W_WUT;
  _Float16* wdt   = w16 + W_WDT;
  _Float16* p_wt  = w16 + W_PT;

  auto blocks = [](long n, int bs) { return (unsigned)((n + bs - 1) / bs); };

  // ---- convert weights to f16 [N, Kpad] ----
  cvt_w_kernel<<<blocks(256L * D0P, 256), 256, 0, stream>>>(s_w0, s_w0t, 56, 256, D0P);
  for (int i = 0; i < 3; ++i)
    cvt_w_kernel<<<blocks(256L * NSVO, 256), 256, 0, stream>>>(s_w + (size_t)i * NSVO * 256,
                                                               s_wt + (size_t)i * 256 * NSVO, NSVO, 256, NSVO);
  cvt_w_kernel<<<blocks(256L * NSVO, 256), 256, 0, stream>>>(vu_w, vut, NSVO, 256, NSVO);
  cvt_w_kernel<<<blocks(256L * NSVO, 256), 256, 0, stream>>>(vd_w, vdt, NSVO, 256, NSVO);
  cvt_w_kernel<<<blocks(160L * 256, 256), 256, 0, stream>>>(wu_w, wut, 256, 160, 256);
  cvt_w_kernel<<<blocks(160L * 256, 256), 256, 0, stream>>>(wd_w, wdt, 256, 160, 256);
  for (int i = 0; i < 3; ++i)
    cvt_w_kernel<<<blocks(32L * 32, 256), 256, 0, stream>>>(p_w + (size_t)i * 32 * 32,
                                                            p_wt + (size_t)i * 32 * 32, 32, 32, 32);

  // ---- features ----
  feat_kernel<<<blocks((long)NB * NE, 256), 256, 0, stream>>>(r, a, sfeat, expv);
  pair0_kernel<<<blocks((long)NB * NE * NE, 256), 256, 0, stream>>>(r, p_w0, p_b0, pv, pfeat);

  // ---- layer 0: materialized concat (56->64), sv = tanh(blk0 @ s_w0) ----
  pmean_kernel<<<blocks((long)NB * NE * 4, 256), 256, 0, stream>>>(pfeat, pmean, 4, NB * NE * 4);
  smean_kernel<<<blocks((long)NB * 16, 256), 256, 0, stream>>>(sfeat, smean, 16, NB * 16);
  blk0_build_kernel<<<blocks((long)NB * NE * D0P, 256), 256, 0, stream>>>(sfeat, smean, pmean, blk0,
                                                                          NB * NE * D0P);
  gemm_tanh_t<4, false><<<dim3(NB * NE / 16, 1), 128, 0, stream>>>(
      blk0, D0P, nullptr, nullptr, nullptr, s_w0t, D0P, s_b0, sv, NSV, nullptr, NSV, D0P, 0, 0);

  // ---- residual layers: fb_block fused into the GEMM A-path ----
  for (int i = 0; i < 3; ++i) {
    pmean_kernel<<<blocks((long)NB * NE * NPV, 256), 256, 0, stream>>>(pv, pmean, NPV, NB * NE * NPV);
    smean_kernel<<<blocks((long)NB * NSV, 256), 256, 0, stream>>>(sv, smean, NSV, NB * NSV);
    gemm_tanh_t<4, true><<<dim3(NB * NE / 16, 1), 128, 0, stream>>>(
        nullptr, 0, sv, smean, pmean, s_wt + (size_t)i * 256 * NSVO, NSVO,
        s_b + (size_t)i * NSV, sv, NSV, sv, NSV, NSVO, 0, 0);
    gemm_tanh_t<1, false><<<dim3(NB * NE * NE / 16, 1), 64, 0, stream>>>(
        pv, NPV, nullptr, nullptr, nullptr, p_wt + (size_t)i * 32 * 32, NPV,
        p_b + (size_t)i * NPV, pv, NPV, pv, NPV, NPV, 0, 0);
  }

  // ---- final fb_block stats, then s_u/s_d via fused A (overwrite sv rows) ----
  pmean_kernel<<<blocks((long)NB * NE * NPV, 256), 256, 0, stream>>>(pv, pmean, NPV, NB * NE * NPV);
  smean_kernel<<<blocks((long)NB * NSV, 256), 256, 0, stream>>>(sv, smean, NSV, NB * NSV);
  gemm_tanh_t<4, true><<<dim3(NB * NU / 16, 1), 128, 0, stream>>>(
      nullptr, 0, sv, smean, pmean, vut, NSVO, vu_b, sv, NSV, nullptr, NSV, NSVO, 1, 0);
  gemm_tanh_t<4, true><<<dim3(NB * NU / 16, 1), 128, 0, stream>>>(
      nullptr, 0, sv, smean, pmean, vdt, NSVO, vd_b, sv, NSV, nullptr, NSV, NSVO, 1, 10);

  // ---- orbitals with envelope ----
  gemm_orb_kernel<<<dim3(NB * NU / 16, 1), 128, 0, stream>>>(sv, NSV, wut, 256, wu_b, expv,
                                                             orb, 0, NDET * NU, 256);
  gemm_orb_kernel<<<dim3(NB * NU / 16, 1), 128, 0, stream>>>(sv, NSV, wdt, 256, wd_b, expv,
                                                             orb + (size_t)NB * NDET * 100, 10, NDET * NU, 256);

  // ---- determinants + combine ----
  det_kernel<<<blocks(2L * NB * NDET, 128), 128, 0, stream>>>(orb, ldv, sgnv, 2 * NB * NDET);
  final_kernel<<<blocks((long)NB, 128), 128, 0, stream>>>(ldv, sgnv, wf_w, out);
}
